// PolyHashV8M_42606075576718
// MI455X (gfx1250) — matrix operand order repaired and checked by reading, not compiled, not run on hardware
//
#include <hip/hip_runtime.h>
#include <hip/hip_bf16.h>

typedef __bf16 bf16_t;
typedef __attribute__((ext_vector_type(16))) __bf16 v16bf;
typedef __attribute__((ext_vector_type(8)))  float    v8f;
typedef __attribute__((ext_vector_type(4)))  unsigned  v4u;

// ---------------- constants ----------------
#define BATCH   8
#define SEQ     4096
#define BT      (BATCH * SEQ)      // 32768
#define E_DIM   32
#define NTAB    16                 // NS*K
#define MASK16  0xFFFFu
#define HF      512
#define FEATP   608                // 580 padded to 19*32
#define HID     1024
#define VOCAB   256

#define ASTRIDE 40                 // A row stride in LDS (32 + 8 pad bf16)
#define BLANE   12                 // dwords per fragment-lane in LDS B (8 + 4 pad)

#if __has_builtin(__builtin_amdgcn_global_load_async_to_lds_b128) && \
    __has_builtin(__builtin_amdgcn_s_wait_asynccnt)
#define HAVE_ASYNC_LDS 1
#else
#define HAVE_ASYNC_LDS 0
#endif

__device__ __constant__ unsigned c_base_primes[16] = {
    2654435761u, 2246822519u, 3266489917u, 668265263u, 374761393u,
    3042594569u, 2428444049u, 1609587929u, 2870177291u, 3203950429u,
    2447445413u, 2862933555u, 3735928559u, 2578838749u, 3580979921u, 1812433253u};
__device__ __constant__ unsigned c_salts[4] = {
    0u, 3735928559u, 3405691582u, 2343432205u};
__device__ __constant__ unsigned c_cond_primes[8] = {
    2654435761u, 2246822519u, 3266489917u, 2028178513u,
    1220703125u, 1610612741u, 805306457u, 402653189u};

// ---------------- F1: hashes + embedding gathers + conditioning ----------------
// one wave (32 lanes) per token; lane = E dim for coalesced gathers
__global__ __launch_bounds__(256)
void f1_hash_gather(const int* __restrict__ chars,
                    const float* __restrict__ tables,
                    const float* __restrict__ cond_w,
                    float* __restrict__ raw) {
  const int lane = threadIdx.x & 31;
  const int wid  = (int)((blockIdx.x * blockDim.x + threadIdx.x) >> 5);
  if (wid >= BT) return;
  const int t = wid & (SEQ - 1);
  const int* crow = chars + (wid & ~(SEQ - 1));

  unsigned key = 0;
  if (lane < 16) {
    const int s = lane >> 2, k = lane & 3;
    const int w = 2 << s;                       // windows 2,4,8,16
    const unsigned salt = c_salts[k];
    unsigned h = 0;
    for (int i = 0; i < w; ++i) {
      int tp = t - (i + 1);
      unsigned tok = (tp >= 0) ? (unsigned)crow[tp] : 0u;
      h ^= tok * (c_base_primes[i] ^ salt);
    }
    key = h & MASK16;
  }

  float emb[NTAB];
  #pragma unroll
  for (int i = 0; i < 8; ++i) {                 // short scales (s=0,1)
    unsigned ki = (unsigned)__shfl((int)key, i, 32);
    emb[i] = tables[(((long)i << 16) + ki) * E_DIM + lane];
  }

  // conditioning logits: short_cat (256) @ cond_w.T (8x256)
  float lg[8];
  #pragma unroll
  for (int j = 0; j < 8; ++j) lg[j] = 0.f;
  #pragma unroll
  for (int i = 0; i < 8; ++i) {
    #pragma unroll
    for (int j = 0; j < 8; ++j)
      lg[j] += emb[i] * cond_w[j * 256 + i * 32 + lane];
  }
  #pragma unroll
  for (int j = 0; j < 8; ++j) {
    #pragma unroll
    for (int off = 16; off > 0; off >>= 1)
      lg[j] += __shfl_xor(lg[j], off, 32);
  }
  unsigned ckey = 0;
  #pragma unroll
  for (int j = 0; j < 8; ++j)
    if (lg[j] > 0.f) ckey ^= c_cond_primes[j];

  #pragma unroll
  for (int i = 8; i < 16; ++i) {                // long scales (s=2,3), salted keys
    unsigned ki = ((unsigned)__shfl((int)key, i, 32) ^ ckey) & MASK16;
    emb[i] = tables[(((long)i << 16) + ki) * E_DIM + lane];
  }

  float* out = raw + (long)wid * HF + lane;
  #pragma unroll
  for (int i = 0; i < NTAB; ++i) out[i * 32] = emb[i];
}

// ---------------- F2: depthwise conv + SiLU + assemble bf16 feature ----------------
__global__ __launch_bounds__(256)
void f2_assemble(const int* __restrict__ chars,
                 const float* __restrict__ byte_table,
                 const float* __restrict__ dw_w,
                 const float* __restrict__ dw_b,
                 const float* __restrict__ raw,
                 bf16_t* __restrict__ feat) {
  long idx = (long)blockIdx.x * blockDim.x + threadIdx.x;
  if (idx >= (long)BT * FEATP) return;
  const int  j   = (int)(idx % FEATP);
  const long tok = idx / FEATP;
  const int  t   = (int)(tok & (SEQ - 1));

  float val = 0.f;
  if (j < 64) {                                     // byte embedding
    val = byte_table[chars[tok] * 64 + j];
  } else if (j < 64 + HF) {                         // causal depthwise conv + SiLU
    const int c = j - 64;
    float acc = dw_b[c];
    #pragma unroll
    for (int q = 0; q < 4; ++q) {
      int tp = t - 3 + q;
      if (tp >= 0) acc += dw_w[c * 4 + q] * raw[(tok - 3 + q) * HF + c];
    }
    val = acc / (1.f + __expf(-acc));
  } else if (j < 64 + HF + 4) {                     // match flags, offsets 1,2,4,8
    const int off = 1 << (j - (64 + HF));
    if (t >= off && chars[tok] == chars[tok - off]) val = 1.f;
  }                                                 // else: zero pad 580..607
  feat[tok * FEATP + j] = (bf16_t)val;
}

// ---------------- weight -> B-fragment swizzle (bf16) ----------------
// frag(kt, nt): 32 lanes x 8 dwords; lane l holds N = nt*16+(l&15),
// K pairs {2v,2v+1} + (l >= 16 ? 16 : 0) + kt*32  (contiguous k order, same as A fragments)
__global__ __launch_bounds__(256)
void wswizzle(const float* __restrict__ W, unsigned* __restrict__ Bsw,
              int N, int Ksrc, int Kpad) {
  const int Ntiles = N >> 4;
  long idx = (long)blockIdx.x * blockDim.x + threadIdx.x;
  long total = (long)(Kpad >> 5) * Ntiles * 256;
  if (idx >= total) return;
  const int v  = (int)(idx & 7);
  const int l  = (int)((idx >> 3) & 31);
  const long f = idx >> 8;
  const int nt = (int)(f % Ntiles);
  const int kt = (int)(f / Ntiles);
  const int n  = nt * 16 + (l & 15);
  const int kb = kt * 32 + ((l & 16) ? 16 : 0) + 2 * v;
  float w0 = (kb     < Ksrc) ? W[(long)n * Ksrc + kb]     : 0.f;
  float w1 = (kb + 1 < Ksrc) ? W[(long)n * Ksrc + kb + 1] : 0.f;
  unsigned short u0 = __builtin_bit_cast(unsigned short, (bf16_t)w0);
  unsigned short u1 = __builtin_bit_cast(unsigned short, (bf16_t)w1);
  Bsw[idx] = (unsigned)u0 | ((unsigned)u1 << 16);
}

// ---------------- WMMA GEMM with async global->LDS double buffering ----------------
// 256 threads = 8 waves; WG tile 128x128; wave tile 32(M) x 64(N) = 2x4 WMMA frags
struct Frag32 { v4u lo, hi; };

#if HAVE_ASYNC_LDS
// builtin prototype (from toolchain diagnostic): pointers to int4 (vector_size 16)
typedef int v4i_vs __attribute__((vector_size(16)));
typedef __attribute__((address_space(1))) v4i_vs* global_v4i_p;
typedef __attribute__((address_space(3))) v4i_vs* lds_v4i_p;

__device__ __forceinline__ void async_cp16(const void* g, void* l) {
  __builtin_amdgcn_global_load_async_to_lds_b128(
      (global_v4i_p)g, (lds_v4i_p)l, 0, 0);
}
#endif

template <int MODE>   // 0: SiLU->bf16   1: res + x + b ->bf16   2: plain ->f32
__global__ __launch_bounds__(256)
void gemm_wmma(const bf16_t* __restrict__ A, int lda,
               const unsigned* __restrict__ Bsw,
               const float* __restrict__ bias,
               void* __restrict__ Cout,
               const bf16_t* __restrict__ Res,
               int N, int Kt) {
  __shared__ __align__(16) bf16_t   Abuf[2][128 * ASTRIDE];
  __shared__ __align__(16) unsigned Bbuf[2][8 * 32 * BLANE];

  const int tid  = threadIdx.x;
  const int lane = tid & 31;
  const int wid  = tid >> 5;
  const int wm   = wid & 3;                 // 4 wave rows (32 M each)
  const int wn   = wid >> 2;                // 2 wave cols (64 N each)
  const int m0   = blockIdx.y * 128;
  const int nblk = blockIdx.x * 128;
  const int Ntiles = N >> 4;

  // ---- copy-side addressing (coalesced; 32B per thread per tile per matrix) ----
  const int ra = tid >> 1;                  // A row 0..127
  const int ka = (tid & 1) * 16;            // elem 0 / 16
  const bf16_t* agsrc = A + (long)(m0 + ra) * lda + ka;
  const int aoff = ra * ASTRIDE + ka;

  const int bjl = tid >> 5;                 // fragment column 0..7 within block
  const int bl  = tid & 31;                 // fragment lane
  const unsigned* bgsrc = Bsw + ((long)(nblk >> 4) + bjl) * 256 + bl * 8;
  const int boff = (bjl * 32 + bl) * BLANE;

  // ---- compute-side LDS offsets ----
  const int krem  = (lane & 16) ? 16 : 0;
  const int aread = (wm * 32 + (lane & 15)) * ASTRIDE + krem;
  const int bread = ((wn * 4) * 32 + lane) * BLANE;

  v8f acc[2][4] = {};

  auto compute_tile = [&](int cur) {
    v16bf afr[2], bfr[4];
    #pragma unroll
    for (int i = 0; i < 2; ++i) {
      const bf16_t* ap = &Abuf[cur][aread + i * 16 * ASTRIDE];
      Frag32 f;
      f.lo = *(const v4u*)(ap);
      f.hi = *(const v4u*)(ap + 8);
      afr[i] = __builtin_bit_cast(v16bf, f);
    }
    #pragma unroll
    for (int j = 0; j < 4; ++j) {
      const unsigned* bp = &Bbuf[cur][bread + j * 32 * BLANE];
      Frag32 f;
      f.lo = *(const v4u*)(bp);
      f.hi = *(const v4u*)(bp + 4);
      bfr[j] = __builtin_bit_cast(v16bf, f);
    }
    #pragma unroll
    for (int j = 0; j < 4; ++j) {
      acc[0][j] = __builtin_amdgcn_wmma_f32_16x16x32_bf16(
          false, afr[0], false, bfr[j], (short)0, acc[0][j], false, false);
      acc[1][j] = __builtin_amdgcn_wmma_f32_16x16x32_bf16(
          false, afr[1], false, bfr[j], (short)0, acc[1][j], false, false);
    }
  };

#if HAVE_ASYNC_LDS
  auto issue_tile = [&](int kt, int buf) {   // 4 async b128 per thread
    const bf16_t* ag = agsrc + kt * 32;
    bf16_t* al = &Abuf[buf][aoff];
    async_cp16(ag,     al);
    async_cp16(ag + 8, al + 8);
    const unsigned* bg = bgsrc + (long)kt * Ntiles * 256;
    unsigned* blp = &Bbuf[buf][boff];
    async_cp16(bg,     blp);
    async_cp16(bg + 4, blp + 4);
  };

  issue_tile(0, 0);
  if (Kt > 1) issue_tile(1, 1);
  for (int kt = 0; kt < Kt; ++kt) {
    const int cur = kt & 1;
    // async ops complete in order: <=4 outstanding means tile kt has landed
    if (kt + 1 < Kt) __builtin_amdgcn_s_wait_asynccnt(4);
    else             __builtin_amdgcn_s_wait_asynccnt(0);
    __syncthreads();
    compute_tile(cur);
    __syncthreads();                         // everyone done reading buf `cur`
    if (kt + 2 < Kt) issue_tile(kt + 2, cur);
  }
#else
  for (int kt = 0; kt < Kt; ++kt) {
    const bf16_t* ag = agsrc + kt * 32;
    v4u a0 = *(const v4u*)(ag);
    v4u a1 = *(const v4u*)(ag + 8);
    const unsigned* bg = bgsrc + (long)kt * Ntiles * 256;
    v4u b0 = *(const v4u*)(bg);
    v4u b1 = *(const v4u*)(bg + 4);
    *(v4u*)(&Abuf[0][aoff])     = a0;
    *(v4u*)(&Abuf[0][aoff + 8]) = a1;
    *(v4u*)(&Bbuf[0][boff])     = b0;
    *(v4u*)(&Bbuf[0][boff + 4]) = b1;
    __syncthreads();
    compute_tile(0);
    __syncthreads();
  }
#endif

  // epilogue: C/D layout: vgpr r, lanes0-15 -> (M=r, N=lane); lanes16-31 -> (M=r+8)
  const int colBase = nblk + wn * 64 + (lane & 15);
  const int rowOff  = (lane & 16) ? 8 : 0;
  const int mwave   = m0 + wm * 32;
  #pragma unroll
  for (int i = 0; i < 2; ++i) {
    #pragma unroll
    for (int j = 0; j < 4; ++j) {
      const int col = colBase + j * 16;
      const float bv = bias[col];
      #pragma unroll
      for (int r = 0; r < 8; ++r) {
        const long row = mwave + i * 16 + rowOff + r;
        float x = acc[i][j][r] + bv;
        if (MODE == 0) {
          float y = x / (1.f + __expf(-x));
          ((bf16_t*)Cout)[row * N + col] = (bf16_t)y;
        } else if (MODE == 1) {
          float y = x + (float)Res[row * N + col];
          ((bf16_t*)Cout)[row * N + col] = (bf16_t)y;
        } else {
          ((float*)Cout)[row * N + col] = x;
        }
      }
    }
  }
}

// ---------------- host orchestration ----------------
extern "C" void kernel_launch(void* const* d_in, const int* in_sizes, int n_in,
                              void* d_out, int out_size, void* d_ws, size_t ws_size,
                              hipStream_t stream) {
  const int*   chars      = (const int*)d_in[0];
  const float* byte_table = (const float*)d_in[1];
  const float* tables     = (const float*)d_in[2];
  const float* cond_w     = (const float*)d_in[3];
  const float* dw_w       = (const float*)d_in[4];
  const float* dw_b       = (const float*)d_in[5];
  const float* in_w       = (const float*)d_in[6];
  const float* in_b       = (const float*)d_in[7];
  const float* res_w1     = (const float*)d_in[8];
  const float* res_b1     = (const float*)d_in[9];
  const float* res_w2     = (const float*)d_in[10];
  const float* res_b2     = (const float*)d_in[11];
  const float* out_w      = (const float*)d_in[12];
  const float* out_b      = (const float*)d_in[13];

  char* ws = (char*)d_ws;
  // raw (f32 BT*512 = 64MB) is dead after f2 -> reuse as tmp (bf16 BT*1024 = 64MB)
  float*  raw  = (float*)(ws);
  bf16_t* tmp  = (bf16_t*)(ws);
  bf16_t* feat = (bf16_t*)(ws + 67108864L);                 // BT*608*2 = 39,845,888
  bf16_t* h    = (bf16_t*)(ws + 67108864L + 39845888L);     // BT*1024*2 = 67,108,864
  unsigned* bsw = (unsigned*)(ws + 174063616L);             // swizzled weights
  unsigned* bsw_in = bsw;                                   // 19*64*256 dwords
  unsigned* bsw_w1[3], *bsw_w2[3];
  unsigned* p = bsw_in + 19L * 64 * 256;
  for (int i = 0; i < 3; ++i) {
    bsw_w1[i] = p; p += 32L * 64 * 256;
    bsw_w2[i] = p; p += 32L * 64 * 256;
  }
  unsigned* bsw_out = p;                                    // 32*16*256 dwords

  // 1) weight swizzles
  auto launch_sw = [&](const float* W, unsigned* B, int N, int Ksrc, int Kpad) {
    long total = (long)(Kpad >> 5) * (N >> 4) * 256;
    wswizzle<<<dim3((unsigned)((total + 255) / 256)), dim3(256), 0, stream>>>(
        W, B, N, Ksrc, Kpad);
  };
  launch_sw(in_w, bsw_in, HID, 580, FEATP);
  for (int i = 0; i < 3; ++i) {
    launch_sw(res_w1 + (long)i * HID * HID, bsw_w1[i], HID, HID, HID);
    launch_sw(res_w2 + (long)i * HID * HID, bsw_w2[i], HID, HID, HID);
  }
  launch_sw(out_w, bsw_out, VOCAB, HID, HID);

  // 2) featurize
  f1_hash_gather<<<dim3(BT / 8), dim3(256), 0, stream>>>(chars, tables, cond_w, raw);
  f2_assemble<<<dim3((unsigned)(((long)BT * FEATP + 255) / 256)), dim3(256), 0, stream>>>(
      chars, byte_table, dw_w, dw_b, raw, feat);

  // 3) MLP via WMMA GEMMs  (grid = (N/128, M/128), M = 32768)
  gemm_wmma<0><<<dim3(HID / 128, BT / 128), dim3(256), 0, stream>>>(
      feat, FEATP, bsw_in, in_b, h, nullptr, HID, FEATP / 32);
  for (int i = 0; i < 3; ++i) {
    gemm_wmma<0><<<dim3(HID / 128, BT / 128), dim3(256), 0, stream>>>(
        h, HID, bsw_w1[i], res_b1 + i * HID, tmp, nullptr, HID, HID / 32);
    gemm_wmma<1><<<dim3(HID / 128, BT / 128), dim3(256), 0, stream>>>(
        tmp, HID, bsw_w2[i], res_b2 + i * HID, h, h, HID, HID / 32);
  }
  gemm_wmma<2><<<dim3(VOCAB / 128, BT / 128), dim3(256), 0, stream>>>(
      h, HID, bsw_out, out_b, d_out, nullptr, VOCAB, HID / 32);
}
